// PSFField_68341519614239
// MI455X (gfx1250) — matrix operand order repaired and checked by reading, not compiled, not run on hardware
//
#include <hip/hip_runtime.h>
#include <hip/hip_bf16.h>
#include <cmath>

typedef __attribute__((ext_vector_type(16))) _Float16 v16h;
typedef __attribute__((ext_vector_type(8)))  _Float16 v8h;
typedef __attribute__((ext_vector_type(8)))  float    v8f;

#define N_SRC   64
#define SS      33
#define HID     128
#define NLAYER  4      // DEPTH-1 hidden 128x128 layers
#define AST     136    // activation row stride in halfs (272B = 17*16 -> aligned, conflict-free)
#define WST     136    // weight row stride in halfs
#define WAVES   8      // waves per block (wave32)
#define PPW     2      // pixels (M=16 tiles) per wave; share B fragments

#define INV2PI  0.15915494309189535f

__device__ __forceinline__ float silu_f(float x) { return x / (1.f + __expf(-x)); }

// ---------------------------------------------------------------------------
// Prep 1: per-source SED MLP, DCR shift, Fourier tile context,
// ctx_pre = (30/2pi) * (ctx @ W0ctx^T + b0)   [pre-scaled for v_sin_f32]
// ---------------------------------------------------------------------------
__global__ __launch_bounds__(128) void psf_prep(
    const float* __restrict__ cen, const float* __restrict__ tile_pos,
    const int*   __restrict__ band_idx, const float* __restrict__ sed_vec,
    const float* __restrict__ band_table, const float* __restrict__ dcr,
    const float* __restrict__ W1, const float* __restrict__ b1,
    const float* __restrict__ W2, const float* __restrict__ b2,
    const float* __restrict__ W3, const float* __restrict__ b3,
    const float* __restrict__ W0, const float* __restrict__ b0,
    float* __restrict__ ctx_pre, float* __restrict__ eff_cen)
{
    __shared__ float sed_s[10], h1[32], h2[32], ctx[48];
    const int n = blockIdx.x, t = threadIdx.x;
    if (t < 10) sed_s[t] = sed_vec[n * 10 + t];
    __syncthreads();
    if (t < 32) {
        float a = b1[t];
        #pragma unroll
        for (int j = 0; j < 10; ++j) a += W1[t * 10 + j] * sed_s[j];
        h1[t] = silu_f(a);
    }
    __syncthreads();
    if (t < 32) {
        float a = b2[t];
        #pragma unroll
        for (int j = 0; j < 32; ++j) a += W2[t * 32 + j] * h1[j];
        h2[t] = silu_f(a);
    }
    __syncthreads();
    if (t < 12) {                       // tile_ctx: sin/cos of 2 coords x 6 freqs
        int d = t / 6, fi = t % 6;
        float f = exp2f(0.6f * (float)fi);            // 8^(i/5) = 2^(3i/5)
        float p = tile_pos[n * 2 + d] * f * 6.28318530717958647692f;
        ctx[t]      = __sinf(p);
        ctx[12 + t] = __cosf(p);
    } else if (t >= 32 && t < 48) {     // band embedding
        int b = band_idx[n];
        ctx[24 + (t - 32)] = band_table[b * 16 + (t - 32)];
    } else if (t >= 64 && t < 72) {     // sed embedding (8)
        int e = t - 64;
        float a = b3[e];
        #pragma unroll
        for (int j = 0; j < 32; ++j) a += W3[e * 32 + j] * h2[j];
        ctx[40 + e] = a;
    }
    __syncthreads();
    {   // pre-scaled: sin(30*z) == v_sin((30/2pi)*z)
        float a = b0[t];
        for (int c = 0; c < 48; ++c) a += W0[t * 50 + 2 + c] * ctx[c];
        ctx_pre[n * HID + t] = a * (30.f * INV2PI);
    }
    if (t < 2) {
        int b = band_idx[n];
        int bc = b < 0 ? 0 : (b > 5 ? 5 : b);
        float color = sed_s[1] - sed_s[3];
        float sh = (b < 6) ? dcr[bc * 2 + t] * color : 0.f;
        eff_cen[n * 2 + t] = cen[n * 2 + t] + sh;
    }
}

// ---------------------------------------------------------------------------
// Prep 2: one-time f32 -> f16 weight image (1/2pi pre-scaled, padded WST rows)
// and pre-scaled biases, into workspace.  65536 weights, 512 biases.
// ---------------------------------------------------------------------------
__global__ __launch_bounds__(256) void psf_wprep(
    const float* __restrict__ Wh, const float* __restrict__ bh,
    _Float16* __restrict__ wgt16, float* __restrict__ bias16)
{
    const int idx = blockIdx.x * 256 + threadIdx.x;     // 0 .. 65535
    if (idx < NLAYER * HID * HID) {
        const int L = idx >> 14;
        const int o = (idx >> 7) & 127;
        const int k = idx & 127;
        wgt16[((size_t)(L * HID + o)) * WST + k] = (_Float16)(Wh[idx] * INV2PI);
    }
    if (idx < NLAYER * HID) bias16[idx] = bh[idx] * INV2PI;
}

// ---------------------------------------------------------------------------
// Main: one wave = two pixels (2 x M=16 sub-sample tiles sharing B fragments).
// 4 dense 128x128 f16 layers via v_wmma_f32_16x16x32_f16, acts in LDS.
// Weights/biases pre-scaled by 1/(2pi) so v_sin_f32 consumes WMMA D directly.
// ---------------------------------------------------------------------------
__global__ __launch_bounds__(256) void psf_main(
    const _Float16* __restrict__ wgt16, const float* __restrict__ bias16,
    const float* __restrict__ siren_W0, const float* __restrict__ head_W,
    const float* __restrict__ head_b,  const float* __restrict__ px_scale_p,
    const float* __restrict__ ctx_pre, const float* __restrict__ eff_cen,
    float* __restrict__ out)
{
    __shared__ _Float16 wgt[HID * WST];               // 34816 B (one layer, f16, /2pi)
    __shared__ _Float16 act[WAVES][PPW][16 * AST];    // 69632 B
    __shared__ float    bias_s[HID];                  // /2pi
    __shared__ float    headw_s[HID];
    __shared__ float    w0xy_s[HID * 2];              // K30-scaled first-layer xy weights

    const int tid  = threadIdx.x;
    const int wave = tid >> 5;
    const int lane = tid & 31;

    const float ps   = px_scale_p[0];
    const float K30  = 30.f * INV2PI;
    if (tid < HID) {
        headw_s[tid]       = head_W[tid];
        w0xy_s[2 * tid]     = K30 * siren_W0[tid * 50 + 0];
        w0xy_s[2 * tid + 1] = K30 * siren_W0[tid * 50 + 1];
    }
    __syncthreads();

    const int   pix0   = (blockIdx.x * WAVES + wave) * PPW;  // global pixel id base
    const float half_s = 0.5f * (float)(SS - 1);

    // ---- first layer per pixel: h0[m][h] = v_sin(x*W0x' + y*W0y' + ctx_pre'[h])
    #pragma unroll
    for (int p = 0; p < PPW; ++p) {
        const int gp  = pix0 + p;
        const int n   = gp / (SS * SS);
        const int pix = gp % (SS * SS);
        const float X = ((float)(pix % SS) - half_s) * ps - eff_cen[n * 2 + 0];
        const float Y = ((float)(pix / SS) - half_s) * ps - eff_cen[n * 2 + 1];
        const float* cpre = ctx_pre + n * HID;
        _Float16* A = act[wave][p];
        for (int idx = lane; idx < 16 * HID; idx += 32) {
            int m = idx >> 7;                      // m = ky*4 + kx
            int h = idx & 127;
            float sx = (((float)(m & 3)  + 0.5f) * 0.25f - 0.5f) * ps;
            float sy = (((float)(m >> 2) + 0.5f) * 0.25f - 0.5f) * ps;
            float pre = (X + sx) * w0xy_s[2 * h] + (Y + sy) * w0xy_s[2 * h + 1] + cpre[h];
            A[m * AST + h] = (_Float16)__builtin_amdgcn_sinf(pre);
        }
    }

    const int mrow  = lane & 15;          // A-fragment row (both lane halves)
    const int kb    = (lane >> 4) * 16;   // A-fragment K sub-base
    const int mbase = (lane >> 4) * 8;    // D-fragment row base
    const int ncol  = lane & 15;          // D/B-fragment column
    const int kgrp  = (lane >> 4) * 16;   // B-fragment K base

    for (int L = 0; L < NLAYER; ++L) {
        __syncthreads();                  // everyone done with previous layer's wgt
        // stage pre-converted f16 weights: straight 16B copy (2176 uint4)
        {
            const uint4* src = (const uint4*)(wgt16 + (size_t)L * HID * WST);
            uint4* dst = (uint4*)wgt;
            for (int i = tid; i < (HID * WST) / 8; i += 256) dst[i] = src[i];
            if (tid < HID) bias_s[tid] = bias16[L * HID + tid];
        }
        __syncthreads();

        // A fragments: PPW tiles x 4 K-chunks (16x32 f16 each)
        v16h afrag[PPW][4];
        #pragma unroll
        for (int p = 0; p < PPW; ++p) {
            const _Float16* Ar = act[wave][p] + mrow * AST + kb;
            #pragma unroll
            for (int kc = 0; kc < 4; ++kc) {
                v8h lo = *(const v8h*)(Ar + kc * 32);        // K = kc*32 + kb + [0..7]
                v8h hi = *(const v8h*)(Ar + kc * 32 + 8);    // + 8
                v16h a;
                #pragma unroll
                for (int j = 0; j < 8; ++j) { a[j] = lo[j]; a[j + 8] = hi[j]; }
                afrag[p][kc] = a;
            }
        }

        #pragma unroll
        for (int nc = 0; nc < 8; ++nc) {
            const float bv = bias_s[nc * 16 + ncol];
            v8f c0, c1;
            #pragma unroll
            for (int r = 0; r < 8; ++r) { c0[r] = bv; c1[r] = bv; }
            #pragma unroll
            for (int kc = 0; kc < 4; ++kc) {
                v16h bfrag = *(const v16h*)(&wgt[(nc * 16 + ncol) * WST + kc * 32 + kgrp]);
                c0 = __builtin_amdgcn_wmma_f32_16x16x32_f16(false, afrag[0][kc], false, bfrag, (short)0, c0, false, false);
                c1 = __builtin_amdgcn_wmma_f32_16x16x32_f16(false, afrag[1][kc], false, bfrag, (short)0, c1, false, false);
            }
            _Float16* s0 = act[wave][0] + mbase * AST + nc * 16 + ncol;
            _Float16* s1 = act[wave][1] + mbase * AST + nc * 16 + ncol;
            #pragma unroll
            for (int r = 0; r < 8; ++r) {   // D: M = mbase+r, N = ncol
                s0[r * AST] = (_Float16)__builtin_amdgcn_sinf(c0[r]);
                s1[r * AST] = (_Float16)__builtin_amdgcn_sinf(c1[r]);
            }
        }
    }

    // ---- head: raw[m] = act[m][:]·head_W + b ; softplus; mean over 16 sub-samples
    const int hbase = (lane >> 4) * 64;
    #pragma unroll
    for (int p = 0; p < PPW; ++p) {
        const _Float16* A = act[wave][p];
        float acc = 0.f;
        for (int j = 0; j < 64; ++j)
            acc += (float)A[mrow * AST + hbase + j] * headw_s[hbase + j];
        acc += __shfl_xor(acc, 16);                   // combine both K halves
        float raw = acc + head_b[0];
        float sp = (raw > 20.f) ? raw : __logf(1.f + __expf(raw));
        #pragma unroll
        for (int off = 1; off < 16; off <<= 1) sp += __shfl_xor(sp, off);
        if (lane == 0) out[pix0 + p] = sp * (1.f / 16.f);
    }
}

// ---------------------------------------------------------------------------
extern "C" void kernel_launch(void* const* d_in, const int* in_sizes, int n_in,
                              void* d_out, int out_size, void* d_ws, size_t ws_size,
                              hipStream_t stream)
{
    const float* cen    = (const float*)d_in[0];
    const float* tpos   = (const float*)d_in[1];
    const int*   bidx   = (const int*)  d_in[2];
    const float* sed    = (const float*)d_in[3];
    // d_in[4] stamp_size (int, =33), d_in[6] sub_grid (int, =4): hardcoded
    const float* pscale = (const float*)d_in[5];
    const float* btab   = (const float*)d_in[7];
    const float* dcr    = (const float*)d_in[8];
    const float* W1 = (const float*)d_in[9],  *b1 = (const float*)d_in[10];
    const float* W2 = (const float*)d_in[11], *b2 = (const float*)d_in[12];
    const float* W3 = (const float*)d_in[13], *b3 = (const float*)d_in[14];
    const float* W0 = (const float*)d_in[15], *b0 = (const float*)d_in[16];
    const float* Wh = (const float*)d_in[17], *bh = (const float*)d_in[18];
    const float* hW = (const float*)d_in[19], *hb = (const float*)d_in[20];

    // workspace layout (all 16B aligned):
    //   [0)        ctx_pre   64*128 f32      = 32768 B
    //   [32768)    eff_cen   64*2  f32       =   512 B
    //   [33280)    bias16    4*128 f32       =  2048 B
    //   [35328)    wgt16     4*128*136 f16   = 139264 B   (total ~174.6 KB)
    char* ws = (char*)d_ws;
    float*     ctx_pre = (float*)    (ws);
    float*     eff_cen = (float*)    (ws + 32768);
    float*     bias16  = (float*)    (ws + 33280);
    _Float16*  wgt16   = (_Float16*) (ws + 35328);

    psf_prep<<<N_SRC, 128, 0, stream>>>(cen, tpos, bidx, sed, btab, dcr,
                                        W1, b1, W2, b2, W3, b3, W0, b0,
                                        ctx_pre, eff_cen);
    psf_wprep<<<256, 256, 0, stream>>>(Wh, bh, wgt16, bias16);

    const int total_pairs = N_SRC * SS * SS / PPW;   // 34848
    const int blocks = total_pairs / WAVES;          // 4356 (exact)
    psf_main<<<blocks, 256, 0, stream>>>(wgt16, bias16, W0, hW, hb, pscale,
                                         ctx_pre, eff_cen, (float*)d_out);
}
